// BDH_48421461295735
// MI455X (gfx1250) — matrix operand order repaired and checked by reading, not compiled, not run on hardware
//
#include <hip/hip_runtime.h>

#define B_     2
#define T_     2048
#define D_     256
#define NH_    4
#define N_     2048
#define VOCAB_ 256
#define NL_    4
#define KNN_   (NH_ * N_)   // 8192

#define USE_ASYNC 1

typedef __attribute__((ext_vector_type(16))) _Float16 v16h;
typedef __attribute__((ext_vector_type(8)))  _Float16 v8h;
typedef __attribute__((ext_vector_type(8)))  float    v8f;

// 16-byte global -> LDS copy. Async path uses the CDNA5 ASYNCcnt-tracked DMA.
__device__ __forceinline__ void cp_g2l_16B(const _Float16* g, _Float16* l) {
#if USE_ASYNC
  unsigned int loff = (unsigned int)(unsigned long long)l;  // flat[31:0] == LDS offset
  asm volatile("global_load_async_to_lds_b128 %0, %1, off"
               :: "v"(loff), "v"(g)
               : "memory");
#else
  *(v8h*)l = *(const v8h*)g;
#endif
}

__device__ __forceinline__ void cp_g2l_wait() {
#if USE_ASYNC
  asm volatile("s_wait_asynccnt 0x0" ::: "memory");
#endif
}

// ---------------------------------------------------------------------------
// f16 WMMA GEMM with double-buffered async LDS staging.
//   C(M x Nn) = A(M x K) * B(K x Nn), B supplied as BT = B^T (Nn x K).
// Block = 256 threads = 8 waves; block tile 64(M) x 128(N); wave tile 16x64.
// K staged 64 halves at a time (two v_wmma_f32_16x16x32_f16 sub-steps per
// accumulator per stage -> 8 WMMAs/wave/stage, one barrier per stage).
// The block DMAs A(64x64) + B(128x64) into LDS once per stage (async),
// while WMMAs consume the previous buffer. Padded LDS stride of 72 halves
// (36 banks) keeps the 16-lane fragment reads bank-conflict-free.
// EPI: 0 = f32 store, 1 = relu->f16, 2 = strict causal mask (col<row)->f16
// Batch z: operand offset = ((z/div)%mod)*stride.
// ---------------------------------------------------------------------------
#define KSTEP  64
#define BSH    72                 // padded LDS stride (halves) for 64-half rows
#define LDS_A  (64 * BSH)         // one A buffer (halves)
#define LDS_B  (128 * BSH)        // one B buffer (halves)

template <int EPI>
__global__ __launch_bounds__(256) void gemm_wmma(
    const _Float16* __restrict__ A, const _Float16* __restrict__ BT,
    void* __restrict__ Cv, int M, int Nn, int K, int lda, int ldb, int ldc,
    long sA, int dA, int mA, long sB, int dB, int mB_, long sC)
{
  (void)M; (void)Nn;
  __shared__ _Float16 smem[2 * LDS_A + 2 * LDS_B];  // 55296 bytes

  const int z = blockIdx.z;
  const _Float16* Ab = A  + ((long)((z / dA) % mA))  * sA;
  const _Float16* Bb = BT + ((long)((z / dB) % mB_)) * sB;
  float*    Cf = (float*)Cv    + (long)z * sC;
  _Float16* Ch = (_Float16*)Cv + (long)z * sC;

  const int tid  = threadIdx.x;
  const int wave = tid >> 5;
  const int lane = tid & 31;
  const int mB = blockIdx.x * 64;    // block row origin
  const int nB = blockIdx.y * 128;   // block col origin
  const int wm = (wave & 3) * 16;    // wave row offset in block tile
  const int wn = (wave >> 2) * 64;   // wave col offset in block tile
  const int lm  = lane & 15;
  const int sel = lane >> 4;

  // Fully-masked block for the scores epilogue (uniform across block:
  // safe to return before any barrier).
  if (EPI == 2 && nB >= mB + 64) {
#pragma unroll
    for (int nt = 0; nt < 4; ++nt) {
      const int gn = nB + wn + nt * 16 + lm;
#pragma unroll
      for (int r = 0; r < 8; ++r) {
        const int gm = mB + wm + sel * 8 + r;
        Ch[(long)gm * ldc + gn] = (_Float16)0.0f;
      }
    }
    return;
  }

  // Staging: A = 64 rows x 8 16B-chunks (512 tasks, 2/thread),
  //          B = 128 cols x 8 16B-chunks (1024 tasks, 4/thread).
  auto stage = [&](int buf, int k) {
    _Float16* la = smem + buf * LDS_A;
    _Float16* lb = smem + 2 * LDS_A + buf * LDS_B;
#pragma unroll
    for (int ps = 0; ps < 2; ++ps) {
      const int id = tid + ps * 256;
      const int r = id >> 3, ch = (id & 7) * 8;
      cp_g2l_16B(Ab + (long)(mB + r) * lda + k + ch, la + r * BSH + ch);
    }
#pragma unroll
    for (int ps = 0; ps < 4; ++ps) {
      const int id = tid + ps * 256;
      const int c = id >> 3, ch = (id & 7) * 8;
      cp_g2l_16B(Bb + (long)(nB + c) * ldb + k + ch, lb + c * BSH + ch);
    }
    if (k + 2 * KSTEP < K)  // warm L2 for the tile after next
      __builtin_prefetch(Ab + (long)(mB + (tid >> 3)) * lda + k + 2 * KSTEP, 0, 1);
  };

  v8f zero = {};
  v8f acc[4] = {zero, zero, zero, zero};

  stage(0, 0);
  cp_g2l_wait();
  __syncthreads();

  for (int k = 0; k < K; k += KSTEP) {
    const int cur = (k / KSTEP) & 1;
    if (k + KSTEP < K) stage(cur ^ 1, k + KSTEP);  // DMA next tile during compute

    const _Float16* labase = smem + cur * LDS_A;
    const _Float16* lbbase = smem + 2 * LDS_A + cur * LDS_B;

#pragma unroll
    for (int ks = 0; ks < KSTEP; ks += 32) {
      // A fragment: lane row = wm+lm, halves [ks+sel*8,+8) and [ks+16+sel*8,+8)
      const _Float16* la = labase + (wm + lm) * BSH + ks + sel * 8;
      v16h af = __builtin_shufflevector(*(const v8h*)la, *(const v8h*)(la + 16),
                                        0, 1, 2, 3, 4, 5, 6, 7,
                                        8, 9, 10, 11, 12, 13, 14, 15);
      // Fetch all 4 B fragments first: in-order LDS returns let the
      // scheduler overlap WMMA nt with the loads for nt+1 (partial dscnt).
      v16h bf[4];
#pragma unroll
      for (int nt = 0; nt < 4; ++nt) {
        const _Float16* lb = lbbase + (wn + nt * 16 + lm) * BSH + ks + sel * 8;
        bf[nt] = __builtin_shufflevector(*(const v8h*)lb, *(const v8h*)(lb + 16),
                                         0, 1, 2, 3, 4, 5, 6, 7,
                                         8, 9, 10, 11, 12, 13, 14, 15);
      }
#pragma unroll
      for (int nt = 0; nt < 4; ++nt) {
        acc[nt] = __builtin_amdgcn_wmma_f32_16x16x32_f16(
            false, af, false, bf[nt], (short)0, acc[nt], false, false);
      }
    }

    cp_g2l_wait();      // our async stores to LDS landed
    __syncthreads();    // everyone's landed; reads of 'cur' complete
  }

#pragma unroll
  for (int nt = 0; nt < 4; ++nt) {
    const int gn = nB + wn + nt * 16 + lm;
#pragma unroll
    for (int r = 0; r < 8; ++r) {
      const int gm = mB + wm + sel * 8 + r;  // C layout: M = (lane/16)*8 + r
      const float v = acc[nt][r];
      if (EPI == 0)       Cf[(long)gm * ldc + gn] = v;
      else if (EPI == 1)  Ch[(long)gm * ldc + gn] = (_Float16)fmaxf(v, 0.0f);
      else                Ch[(long)gm * ldc + gn] = (_Float16)((gn < gm) ? v : 0.0f);
    }
  }
}

// ---------------------------------------------------------------------------
// Helpers / elementwise kernels
// ---------------------------------------------------------------------------
__device__ __forceinline__ float block_sum_256(float v, float* red, int tid) {
  red[tid] = v;
  __syncthreads();
#pragma unroll
  for (int s = 128; s > 0; s >>= 1) {
    if (tid < s) red[tid] += red[tid + s];
    __syncthreads();
  }
  float r = red[0];
  __syncthreads();
  return r;
}

// batched transpose + f32->f16: in (batch, R, C) -> out (batch, C, R)
__global__ void transpose_to_f16(const float* __restrict__ in,
                                 _Float16* __restrict__ out,
                                 int R, int C, long total) {
  long i = (long)blockIdx.x * blockDim.x + threadIdx.x;
  if (i >= total) return;
  long rc  = (long)R * C;
  long b   = i / rc;
  long rem = i - b * rc;
  int  r   = (int)(rem / C);
  int  c   = (int)(rem - (long)r * C);
  out[b * rc + (long)c * R + r] = (_Float16)in[i];
}

// x = ln(embed[idx]); writes x32 (B,T,D), xh f16 (B,T,D), xhT f16 (B,D,T)
__global__ __launch_bounds__(256) void embed_ln(
    const int* __restrict__ idx, const float* __restrict__ embed,
    float* __restrict__ x32, _Float16* __restrict__ xh,
    _Float16* __restrict__ xhT) {
  __shared__ float red[256];
  const int row = blockIdx.x;        // b*T + t
  const int b = row / T_, t = row % T_;
  const int d = threadIdx.x;
  const int tok = idx[row];
  float v = embed[(long)tok * D_ + d];
  float mu = block_sum_256(v, red, d) * (1.0f / D_);
  float c = v - mu;
  float var = block_sum_256(c * c, red, d) * (1.0f / D_);
  float y = c * rsqrtf(var + 1e-5f);
  x32[(long)row * D_ + d] = y;
  xh [(long)row * D_ + d] = (_Float16)y;
  xhT[((long)b * D_ + d) * T_ + t] = (_Float16)y;
}

// RoPE over x_sparse (B*NH*T, N): one thread per even/odd pair
__global__ void rope_k(const _Float16* __restrict__ xsp,
                       _Float16* __restrict__ qr) {
  long i = (long)blockIdx.x * blockDim.x + threadIdx.x;  // over B*NH*T*(N/2)
  const long PP = N_ / 2;
  long row = i / PP;                  // (b,h,t) flat
  int  t   = (int)(row % T_);
  int  n   = (int)(i - row * PP) * 2; // even index; q = floor(n/2)*2 = n
  float v0 = (float)xsp[row * N_ + n];
  float v1 = (float)xsp[row * N_ + n + 1];
  // freqs = theta^(-q/N) / (2*pi), theta = 2^16  ->  exp2(-16*q/N)/(2*pi)
  float freq = exp2f(-16.0f * (float)n / (float)N_) * 0.15915494309189535f;
  float ph = (float)t * freq;
  ph -= floorf(ph);                   // phases % 1.0
  float ang = ph * 6.283185307179586f;
  float s, c;
  __sincosf(ang, &s, &c);
  qr[row * N_ + n]     = (_Float16)(v0 * c - v1 * s);
  qr[row * N_ + n + 1] = (_Float16)(v1 * c + v0 * s);
}

// row-wise LN (rows of length D) f32 -> f16
__global__ __launch_bounds__(256) void ln_rows_f16(
    const float* __restrict__ in, _Float16* __restrict__ out) {
  __shared__ float red[256];
  const int row = blockIdx.x;
  const int d = threadIdx.x;
  float v = in[(long)row * D_ + d];
  float mu = block_sum_256(v, red, d) * (1.0f / D_);
  float c = v - mu;
  float var = block_sum_256(c * c, red, d) * (1.0f / D_);
  out[(long)row * D_ + d] = (_Float16)(c * rsqrtf(var + 1e-5f));
}

// xy[b,t,h*N+n] = x_sparse[b,h,t,n] * y_sparse[b,h,t,n]
__global__ void xy_mul(const _Float16* __restrict__ xsp,
                       const _Float16* __restrict__ ysp,
                       _Float16* __restrict__ xy) {
  long i = (long)blockIdx.x * blockDim.x + threadIdx.x;  // over B*NH*T*N
  long n = i % N_;
  long r = i / N_;            // (b,h,t)
  int  t = (int)(r % T_);
  long bh = r / T_;
  int  h = (int)(bh % NH_);
  int  b = (int)(bh / NH_);
  float v = (float)xsp[i] * (float)ysp[i];
  xy[(((long)b * T_ + t) * NH_ + h) * N_ + n] = (_Float16)v;
}

// x = ln(x + dec); updates x32 in place, refreshes xh and xhT
__global__ __launch_bounds__(256) void residual_ln(
    float* __restrict__ x32, const float* __restrict__ dec,
    _Float16* __restrict__ xh, _Float16* __restrict__ xhT) {
  __shared__ float red[256];
  const int row = blockIdx.x;        // b*T + t
  const int b = row / T_, t = row % T_;
  const int d = threadIdx.x;
  float v = x32[(long)row * D_ + d] + dec[(long)row * D_ + d];
  float mu = block_sum_256(v, red, d) * (1.0f / D_);
  float c = v - mu;
  float var = block_sum_256(c * c, red, d) * (1.0f / D_);
  float y = c * rsqrtf(var + 1e-5f);
  x32[(long)row * D_ + d] = y;
  xh [(long)row * D_ + d] = (_Float16)y;
  xhT[((long)b * D_ + d) * T_ + t] = (_Float16)y;
}

// ---------------------------------------------------------------------------
extern "C" void kernel_launch(void* const* d_in, const int* in_sizes, int n_in,
                              void* d_out, int out_size, void* d_ws, size_t ws_size,
                              hipStream_t stream) {
  (void)in_sizes; (void)n_in; (void)out_size; (void)ws_size;
  const int*   idx       = (const int*)d_in[0];
  const float* embed     = (const float*)d_in[1];
  const float* encoder   = (const float*)d_in[2];
  const float* encoder_v = (const float*)d_in[3];
  const float* decoder   = (const float*)d_in[4];
  const float* lm_head   = (const float*)d_in[5];
  float* out = (float*)d_out;

  char* p = (char*)d_ws;
  auto carve = [&](size_t bytes) -> char* {
    char* r = p;
    p += (bytes + 255) & ~(size_t)255;
    return r;
  };
  float*    x32      = (float*)   carve((size_t)B_ * T_ * D_ * 4);
  _Float16* xh       = (_Float16*)carve((size_t)B_ * T_ * D_ * 2);
  _Float16* xhT      = (_Float16*)carve((size_t)B_ * D_ * T_ * 2);
  _Float16* encT     = (_Float16*)carve((size_t)NH_ * N_ * D_ * 2);
  _Float16* encvT    = (_Float16*)carve((size_t)NH_ * N_ * D_ * 2);
  _Float16* decT     = (_Float16*)carve((size_t)D_ * KNN_ * 2);
  _Float16* lmT      = (_Float16*)carve((size_t)VOCAB_ * D_ * 2);
  _Float16* xsp16    = (_Float16*)carve((size_t)B_ * NH_ * T_ * N_ * 2);
  _Float16* qr16     = (_Float16*)carve((size_t)B_ * NH_ * T_ * N_ * 2);
  _Float16* scores16 = (_Float16*)carve((size_t)B_ * NH_ * T_ * T_ * 2);
  float*    ykv32    = (float*)   carve((size_t)B_ * NH_ * T_ * D_ * 4);
  _Float16* ykv16    = (_Float16*)carve((size_t)B_ * NH_ * T_ * D_ * 2);
  _Float16* ysp16    = (_Float16*)carve((size_t)B_ * NH_ * T_ * N_ * 2);
  _Float16* xy16     = (_Float16*)carve((size_t)B_ * T_ * KNN_ * 2);
  float*    dec32    = (float*)   carve((size_t)B_ * T_ * D_ * 4);

  // --- weight conversion (transposed, f16) ---
  {
    long tot = (long)NH_ * D_ * N_;
    transpose_to_f16<<<(int)((tot + 255) / 256), 256, 0, stream>>>(encoder, encT, D_, N_, tot);
    transpose_to_f16<<<(int)((tot + 255) / 256), 256, 0, stream>>>(encoder_v, encvT, D_, N_, tot);
    long totd = (long)KNN_ * D_;
    transpose_to_f16<<<(int)((totd + 255) / 256), 256, 0, stream>>>(decoder, decT, KNN_, D_, totd);
    long totl = (long)D_ * VOCAB_;
    transpose_to_f16<<<(int)((totl + 255) / 256), 256, 0, stream>>>(lm_head, lmT, D_, VOCAB_, totl);
  }

  // --- x = ln(embed[idx]) ---
  embed_ln<<<B_ * T_, 256, 0, stream>>>(idx, embed, x32, xh, xhT);

  for (int l = 0; l < NL_; ++l) {
    // x_sparse = relu(x @ encoder)   (batch z = b*NH + h)
    dim3 gEnc(T_ / 64, N_ / 128, B_ * NH_);
    gemm_wmma<1><<<gEnc, 256, 0, stream>>>(
        xh, encT, xsp16, T_, N_, D_, D_, D_, N_,
        (long)T_ * D_, NH_, B_,        // A: x per batch b = z/NH
        (long)N_ * D_, 1, NH_,         // B: encoder per head h = z%NH
        (long)T_ * N_);

    // qr = rope(x_sparse)
    rope_k<<<(B_ * NH_ * T_ * (N_ / 2)) / 256, 256, 0, stream>>>(xsp16, qr16);

    // scores = (qr qr^T) * strict_causal_mask
    dim3 gSc(T_ / 64, T_ / 128, B_ * NH_);
    gemm_wmma<2><<<gSc, 256, 0, stream>>>(
        qr16, qr16, scores16, T_, T_, N_, N_, N_, T_,
        (long)T_ * N_, 1, B_ * NH_,
        (long)T_ * N_, 1, B_ * NH_,
        (long)T_ * T_);

    // ykv = ln(scores @ x)
    dim3 gKv(T_ / 64, D_ / 128, B_ * NH_);
    gemm_wmma<0><<<gKv, 256, 0, stream>>>(
        scores16, xhT, ykv32, T_, D_, T_, T_, T_, D_,
        (long)T_ * T_, 1, B_ * NH_,
        (long)D_ * T_, NH_, B_,        // B: x^T per batch b = z/NH
        (long)T_ * D_);
    ln_rows_f16<<<B_ * NH_ * T_, 256, 0, stream>>>(ykv32, ykv16);

    // y_sparse = relu(ykv @ encoder_v)
    gemm_wmma<1><<<gEnc, 256, 0, stream>>>(
        ykv16, encvT, ysp16, T_, N_, D_, D_, D_, N_,
        (long)T_ * D_, 1, B_ * NH_,
        (long)N_ * D_, 1, NH_,
        (long)T_ * N_);

    // xy = (x_sparse * y_sparse) transposed to (B, T, NH*N)
    xy_mul<<<(int)(((long)B_ * NH_ * T_ * N_) / 256), 256, 0, stream>>>(xsp16, ysp16, xy16);

    // dec = xy @ decoder
    dim3 gDec(T_ / 64, D_ / 128, B_);
    gemm_wmma<0><<<gDec, 256, 0, stream>>>(
        xy16, decT, dec32, T_, D_, KNN_, KNN_, KNN_, D_,
        (long)T_ * KNN_, 1, B_,
        0L, 1, 1,                      // decoder shared
        (long)T_ * D_);

    // x = ln(x + dec)
    residual_ln<<<B_ * T_, 256, 0, stream>>>(x32, dec32, xh, xhT);
  }

  // logits = x @ lm_head  (treat (B,T,D) as (B*T, D))
  dim3 gLm((B_ * T_) / 64, VOCAB_ / 128, 1);
  gemm_wmma<0><<<gLm, 256, 0, stream>>>(
      xh, lmT, out, B_ * T_, VOCAB_, D_, D_, D_, VOCAB_,
      0L, 1, 1, 0L, 1, 1, 0L);
}
